// FraudGNN_31963146616897
// MI455X (gfx1250) — matrix operand order repaired and checked by reading, not compiled, not run on hardware
//
#include <hip/hip_runtime.h>
#include <hip/hip_bf16.h>

#define N_TX   500000
#define N_ACC  100000
#define NEDGE  1000000
#define D_TX   64
#define D_ACC  32
#define HDIM   64

typedef __attribute__((ext_vector_type(16))) _Float16 v16h;
typedef __attribute__((ext_vector_type(8)))  float    v8f;

// ---------------------------------------------------------------------------
// Kernel 1: zero agg-sum [N_TX x 32] and count [N_TX] in workspace
// ---------------------------------------------------------------------------
__global__ void zero_ws_kernel(float* __restrict__ p, int n) {
    int i = blockIdx.x * blockDim.x + threadIdx.x;
    if (i < n) p[i] = 0.0f;
}

// ---------------------------------------------------------------------------
// Kernel 2: scatter-add of x_acc[src] into agg[dst], one wave per edge,
// lane = feature (D_ACC == 32 == warpSize). Count bumped by lane 0.
// agg is 64 MB -> L2-resident on MI455X (192 MB L2), so the atomics stay
// on-chip.
// ---------------------------------------------------------------------------
__global__ void scatter_kernel(const float* __restrict__ x_acc,
                               const int*   __restrict__ src,
                               const int*   __restrict__ dst,
                               float* __restrict__ agg,
                               float* __restrict__ cnt) {
    const int lane = threadIdx.x & 31;
    const int e    = blockIdx.x * (blockDim.x >> 5) + (threadIdx.x >> 5);
    if (e >= NEDGE) return;
    const int s = src[e];
    const int d = dst[e];
    const float v = x_acc[(size_t)s * D_ACC + lane];
    atomicAdd(&agg[(size_t)d * D_ACC + lane], v);
    if (lane == 0) atomicAdd(&cnt[d], 1.0f);
}

// ---------------------------------------------------------------------------
// WMMA operand loaders (wave32, 16x16x32 f16 shapes; this code's contiguous K order, same in A and B)
// A 16x32 (MxK): lane l holds row m=l&15; hi=l>>4 selects K-halves:
//   vgpr j   (j<4): K = hi*16 + 2j, 2j+1
//   vgpr 4+j (j<4): K = hi*16 + 8 + 2j, 2j+1
// ---------------------------------------------------------------------------
__device__ inline v16h load_a_f16(const float* __restrict__ rowp, float scale, int hi) {
    v16h a;
#pragma unroll
    for (int j = 0; j < 4; ++j) {
        const int k0 = hi * 16 + 2 * j;
        const int k1 = hi * 16 + 8 + 2 * j;
        a[2 * j]     = (_Float16)(rowp[k0]     * scale);
        a[2 * j + 1] = (_Float16)(rowp[k0 + 1] * scale);
        a[8 + 2 * j]     = (_Float16)(rowp[k1]     * scale);
        a[8 + 2 * j + 1] = (_Float16)(rowp[k1 + 1] * scale);
    }
    return a;
}

// B 32x16 (KxN), W row-major [K][64], n = tile-col-base + (lane&15):
//   vgpr j: K = hi*16 + 2j, 2j+1   (lanes 0-15 K=0..15, lanes 16-31 K=16..31)
__device__ inline v16h load_b_f16(const float* __restrict__ W, int n, int hi) {
    v16h b;
#pragma unroll
    for (int j = 0; j < 8; ++j) {
        const int k = hi * 16 + 2 * j;
        b[2 * j]     = (_Float16)W[(size_t)k * HDIM + n];
        b[2 * j + 1] = (_Float16)W[(size_t)(k + 1) * HDIM + n];
    }
    return b;
}

// ---------------------------------------------------------------------------
// Kernel 3: fused  h = (agg/deg)@Wl + bl + x_tx@Wr ; tx_x = relu(h);
//                  logits = tx_x @ W_out + b_out
// One wave per 16-row tile of tx nodes (500000/16 = 31250 tiles exactly).
// 12 v_wmma_f32_16x16x32_f16 per tile; h never touches memory.
// ---------------------------------------------------------------------------
__global__ void __launch_bounds__(256)
sage_tx_kernel(const float* __restrict__ x_tx,
               const float* __restrict__ agg,
               const float* __restrict__ cnt,
               const float* __restrict__ Wl,    // 32 x 64
               const float* __restrict__ bl,    // 64
               const float* __restrict__ Wr,    // 64 x 64
               const float* __restrict__ W_out, // 64
               const float* __restrict__ b_out, // 1
               float* __restrict__ logits,      // N_TX
               float* __restrict__ tx_x) {      // N_TX x 64
    const int lane = threadIdx.x & 31;
    const int wave = threadIdx.x >> 5;
    const int tile = blockIdx.x * (blockDim.x >> 5) + wave;
    if (tile >= N_TX / 16) return;   // whole-wave early out; active waves keep EXEC all-1s

    const int m    = lane & 15;
    const int hi   = lane >> 4;
    const int row0 = tile * 16;
    const int row  = row0 + m;

    // A operands (shared across all 4 N-tiles)
    const float deg  = cnt[row];
    const float inv  = 1.0f / fmaxf(deg, 1.0f);          // segment-mean scale
    const v16h aAgg  = load_a_f16(agg  + (size_t)row * D_ACC, inv,  hi);
    const v16h aX0   = load_a_f16(x_tx + (size_t)row * D_TX,       1.0f, hi);
    const v16h aX1   = load_a_f16(x_tx + (size_t)row * D_TX + 32,  1.0f, hi);

    float part[8] = {0.f, 0.f, 0.f, 0.f, 0.f, 0.f, 0.f, 0.f};

#pragma unroll
    for (int t = 0; t < 4; ++t) {
        const int n = t * 16 + (lane & 15);

        const v16h bWl  = load_b_f16(Wl, n, hi);              // K = 0..31
        const v16h bWr0 = load_b_f16(Wr, n, hi);              // K = 0..31
        const v16h bWr1 = load_b_f16(Wr + 32 * HDIM, n, hi);  // K = 32..63

        // C init = bias broadcast (C/D: lane n=l&15, vgpr r -> row hi*8+r)
        v8f acc;
        const float bias = bl[n];
#pragma unroll
        for (int r = 0; r < 8; ++r) acc[r] = bias;

        acc = __builtin_amdgcn_wmma_f32_16x16x32_f16(false, aAgg, false, bWl,
                                                     (short)0, acc, false, false);
        acc = __builtin_amdgcn_wmma_f32_16x16x32_f16(false, aX0,  false, bWr0,
                                                     (short)0, acc, false, false);
        acc = __builtin_amdgcn_wmma_f32_16x16x32_f16(false, aX1,  false, bWr1,
                                                     (short)0, acc, false, false);

        // fused epilogue: ReLU, store tx_x, accumulate W_out dot
        const float wo = W_out[n];
#pragma unroll
        for (int r = 0; r < 8; ++r) {
            const float v = fmaxf(acc[r], 0.0f);
            const int mm = hi * 8 + r;
            tx_x[(size_t)(row0 + mm) * HDIM + n] = v;   // 16 lanes -> 64B contiguous
            part[r] += v * wo;
        }
    }

    // reduce partials over the 16 lanes (n = 0..15) of each half-wave
#pragma unroll
    for (int r = 0; r < 8; ++r) {
        float s = part[r];
#pragma unroll
        for (int off = 1; off < 16; off <<= 1)
            s += __shfl_xor(s, off, 32);
        if ((lane & 15) == 0)
            logits[row0 + hi * 8 + r] = s + b_out[0];
    }
}

// ---------------------------------------------------------------------------
// Launcher
// ---------------------------------------------------------------------------
extern "C" void kernel_launch(void* const* d_in, const int* in_sizes, int n_in,
                              void* d_out, int out_size, void* d_ws, size_t ws_size,
                              hipStream_t stream) {
    (void)in_sizes; (void)n_in; (void)out_size; (void)ws_size;

    const float* x_tx   = (const float*)d_in[0];
    const float* x_acc  = (const float*)d_in[1];
    const int*   p_src  = (const int*)  d_in[2];
    const int*   p_dst  = (const int*)  d_in[3];
    // d_in[4], d_in[5]: rev edges  — dead code (h_acc unused downstream)
    const float* Wl     = (const float*)d_in[6];
    const float* bl     = (const float*)d_in[7];
    const float* Wr     = (const float*)d_in[8];
    // d_in[9..11]: Wl_rev / bl_rev / Wr_rev — dead code
    const float* W_out  = (const float*)d_in[12];
    const float* b_out  = (const float*)d_in[13];

    float* logits = (float*)d_out;            // [N_TX]
    float* tx_x   = (float*)d_out + N_TX;     // [N_TX x 64]

    float* agg = (float*)d_ws;                // [N_TX x 32]  (64 MB)
    float* cnt = agg + (size_t)N_TX * D_ACC;  // [N_TX]       ( 2 MB)

    // 1) zero scatter buffers
    {
        const int n = N_TX * D_ACC + N_TX;
        zero_ws_kernel<<<(n + 255) / 256, 256, 0, stream>>>(agg, n);
    }
    // 2) edge scatter (8 edges / 256-thread block, one wave per edge)
    scatter_kernel<<<(NEDGE + 7) / 8, 256, 0, stream>>>(x_acc, p_src, p_dst, agg, cnt);

    // 3) fused WMMA GEMM + bias + ReLU + output head
    {
        const int tiles = N_TX / 16;               // 31250
        const int waves_per_block = 256 / 32;      // 8
        const int blocks = (tiles + waves_per_block - 1) / waves_per_block;
        sage_tx_kernel<<<blocks, 256, 0, stream>>>(x_tx, agg, cnt, Wl, bl, Wr,
                                                   W_out, b_out, logits, tx_x);
    }
}
